// Encoder_61744449847593
// MI455X (gfx1250) — matrix operand order repaired and checked by reading, not compiled, not run on hardware
//
#include <hip/hip_runtime.h>

// ---------------------------------------------------------------------------
// ST-GCN-ish encoder for MI455X (gfx1250, wave32, WMMA).
// Kernels on one stream:
//   0) k_packB     : pre-pack ALL weight matrices into per-lane WMMA B-fragment
//                    layout (f16) in d_ws -> compute waves load b128, no cvt.
//   1) k_upsample1 : agg1 = x @ Wu1 + bu1                       (WMMA)
//   2) k_edge1     : agg1[dst] += sig(z@Wf1+bf1)*sp(z@Ws1+bs1)  (WMMA + atomics)
//   3) k_upsample2 : agg2 = agg1 @ Wu2 + bu2                    (WMMA)
//   4) k_edge2     : agg2[dst] += sig(z@Wf2+bf2)*sp(z@Ws2+bs2)  (WMMA + atomics)
//   5) k_tconv     : out = relu(conv1d_k3_edgepad(agg2)+cb) + agg2 (im2col WMMA)
// All GEMMs: v_wmma_f32_16x16x32_f16 (f16 inputs, f32 accumulate).
// ---------------------------------------------------------------------------

#define H1   16
#define H2   32
#define CIN  8
#define EDIM 4

typedef __attribute__((ext_vector_type(16))) _Float16 v16h;
typedef __attribute__((ext_vector_type(8)))  float    v8f;

// ---- B-fragment table in workspace (each fragment = 32 lanes * 16 halfs) ----
#define FRAG_HALFS   512                      // halfs per fragment
#define FRAG_WU1     0
#define FRAG_WF1     1
#define FRAG_WS1     2
#define FRAG_WU2(t)       (3 + (t))
#define FRAG_WF2(kb, t)   (5 + (t) * 2 + (kb))
#define FRAG_WS2(kb, t)   (9 + (t) * 2 + (kb))
#define FRAG_CW(dk, t)    (13 + (t) * 3 + (dk))
#define N_FRAGS      19
#define PACKB_BYTES  (32768)                  // 19 KB used, padded for alignment

static __device__ __forceinline__ v8f wmma16(v16h a, v16h b, v8f c) {
  // (neg_a, A, neg_b, B, c_mod, C, reuse_a, reuse_b)
  return __builtin_amdgcn_wmma_f32_16x16x32_f16(false, a, false, b, (short)0, c,
                                                false, false);
}

static __device__ __forceinline__ v16h load_frag(const _Float16* packB,
                                                 int frag, int lane) {
  return *(const v16h*)(packB + (size_t)frag * FRAG_HALFS + lane * 16);
}

// Load 8 consecutive f32 (16B-aligned) into a[o..o+7] as f16 via 2x b128 loads.
static __device__ __forceinline__ void load8h(const float* __restrict__ p,
                                              v16h& a, int o) {
  const float4 q0 = ((const float4*)p)[0];
  const float4 q1 = ((const float4*)p)[1];
  a[o + 0] = (_Float16)q0.x; a[o + 1] = (_Float16)q0.y;
  a[o + 2] = (_Float16)q0.z; a[o + 3] = (_Float16)q0.w;
  a[o + 4] = (_Float16)q1.x; a[o + 5] = (_Float16)q1.y;
  a[o + 6] = (_Float16)q1.z; a[o + 7] = (_Float16)q1.w;
}

static __device__ __forceinline__ float sigm(float x) {
  return 1.0f / (1.0f + __expf(-x));
}
static __device__ __forceinline__ float softplusf(float x) {
  return fmaxf(x, 0.0f) + __logf(1.0f + __expf(-fabsf(x)));
}

// ----------------------------------------------------------------------------
// Kernel 0: pack every weight into the ISA B-fragment lane layout:
// B (KxN, 16-bit): lane L holds column N = L&15; half h maps to
// K = (h>=8 ? 16 : 0) + (L>=16 ? 8 : 0) + (h&7) within the 32-K block (plus kb*32 for K blocks).
// ----------------------------------------------------------------------------
extern "C" __global__ void __launch_bounds__(640)
k_packB(const float* __restrict__ Wu1, const float* __restrict__ Wf1,
        const float* __restrict__ Ws1, const float* __restrict__ Wu2,
        const float* __restrict__ Wf2, const float* __restrict__ Ws2,
        const float* __restrict__ cw, _Float16* __restrict__ packB) {
  const int tid = (int)threadIdx.x;
  if (tid >= N_FRAGS * 32) return;
  const int frag = tid >> 5;
  const int lane = tid & 31;
  const int hi = lane >> 4, r = lane & 15;

  v16h out = {};
  for (int h = 0; h < 16; ++h) {
    float v = 0.0f;
    if (frag == FRAG_WU1) {                       // Wu1: 8 x 16
      const int k = (h & 8) * 2 + hi * 8 + (h & 7);
      if (k < CIN) v = Wu1[k * H1 + r];
    } else if (frag == FRAG_WF1 || frag == FRAG_WS1) {  // 20 x 16
      const int k = (h & 8) * 2 + hi * 8 + (h & 7);
      const float* W = (frag == FRAG_WF1) ? Wf1 : Ws1;
      if (k < 2 * CIN + EDIM) v = W[k * H1 + r];
    } else if (frag <= FRAG_WU2(1)) {             // Wu2: 16 x 32, N-tile t
      const int t = frag - FRAG_WU2(0);
      const int k = (h & 8) * 2 + hi * 8 + (h & 7);
      if (k < H1) v = Wu2[k * H2 + t * 16 + r];
    } else if (frag <= FRAG_WS2(1, 1)) {          // Wf2/Ws2: 36 x 32
      const bool isF = frag <= FRAG_WF2(1, 1);
      const int q = frag - (isF ? FRAG_WF2(0, 0) : FRAG_WS2(0, 0));
      const int kb = q & 1, t = q >> 1;
      const int k = kb * 32 + (h & 8) * 2 + hi * 8 + (h & 7);
      const float* W = isF ? Wf2 : Ws2;
      if (k < 2 * H1 + EDIM) v = W[k * H2 + t * 16 + r];
    } else {                                      // conv_w[c][i][dk] (32,32,3)
      const int q = frag - FRAG_CW(0, 0);
      const int dk = q % 3, t = q / 3;
      const int i = (h & 8) * 2 + hi * 8 + (h & 7); // input channel = K
      v = cw[(size_t)(t * 16 + r) * 96 + i * 3 + dk];
    }
    out[h] = (_Float16)v;
  }
  *(v16h*)(packB + (size_t)frag * FRAG_HALFS + lane * 16) = out;
}

// ----------------------------------------------------------------------------
// Kernel 1: agg1[N,16] = x[N,8] @ Wu1[8,16] + bu1   (one 16-row tile per wave)
// ----------------------------------------------------------------------------
extern "C" __global__ void __launch_bounds__(256)
k_upsample1(const float* __restrict__ x, const _Float16* __restrict__ packB,
            const float* __restrict__ bu, float* __restrict__ agg, int nNodes) {
  const int wave = (int)((blockIdx.x * blockDim.x + threadIdx.x) >> 5);
  const int lane = (int)(threadIdx.x & 31);
  const int hi = lane >> 4, r = lane & 15;
  const int base = wave * 16;
  if (base >= nNodes) return;

  v16h a = {};
  if (hi == 0)  // lanes 0-15 carry K=0..7
    load8h(x + (size_t)(base + r) * CIN, a, 0);
  const v16h b = load_frag(packB, FRAG_WU1, lane);
  v8f c = {};
  c = wmma16(a, b, c);

  const float bias = bu[r];
#pragma unroll
  for (int v = 0; v < 8; ++v) {
    const int m = v + hi * 8;
    agg[(size_t)(base + m) * H1 + r] = c[v] + bias;
  }
}

// ----------------------------------------------------------------------------
// Kernel 2: layer-1 edge messages. Per wave: 16 edges.
// z[20] = {x[dst](8), x[src](8), edge_attr(4)}, K padded to 32.
// ----------------------------------------------------------------------------
extern "C" __global__ void __launch_bounds__(256)
k_edge1(const float* __restrict__ x, const float* __restrict__ ea,
        const _Float16* __restrict__ packB,
        const float* __restrict__ bf, const float* __restrict__ bs,
        const int* __restrict__ ei, float* __restrict__ agg, int nEdges) {
  const int wave = (int)((blockIdx.x * blockDim.x + threadIdx.x) >> 5);
  const int lane = (int)(threadIdx.x & 31);
  const int hi = lane >> 4, r = lane & 15;
  const int base = wave * 16;
  if (base >= nEdges) return;
  const int* __restrict__ srcI = ei;           // edge_index[0]
  const int* __restrict__ dstI = ei + nEdges;  // edge_index[1]
  const int e = base + r;

  // A: lanes<16: K 0..7 = x[dst], K 16..19 = ea; lanes>=16: K 8..15 = x[src].
  v16h a = {};
  if (hi == 0) {
    load8h(x + (size_t)dstI[e] * CIN, a, 0);
    const float4 q = *(const float4*)(ea + (size_t)e * EDIM);
    a[8]  = (_Float16)q.x; a[9]  = (_Float16)q.y;
    a[10] = (_Float16)q.z; a[11] = (_Float16)q.w;
  } else {
    load8h(x + (size_t)srcI[e] * CIN, a, 0);
  }

  const v16h vbf = load_frag(packB, FRAG_WF1, lane);
  const v16h vbs = load_frag(packB, FRAG_WS1, lane);
  v8f cf = {}, cs = {};
  cf = wmma16(a, vbf, cf);
  cs = wmma16(a, vbs, cs);

  const float bf0 = bf[r], bs0 = bs[r];
#pragma unroll
  for (int v = 0; v < 8; ++v) {
    const int m = v + hi * 8;
    const int dd = dstI[base + m];
    const float val = sigm(cf[v] + bf0) * softplusf(cs[v] + bs0);
    atomicAdd(&agg[(size_t)dd * H1 + r], val);
  }
}

// ----------------------------------------------------------------------------
// Kernel 3: agg2[N,32] = agg1[N,16] @ Wu2[16,32] + bu2  (2 N-tiles per wave)
// ----------------------------------------------------------------------------
extern "C" __global__ void __launch_bounds__(256)
k_upsample2(const float* __restrict__ h1, const _Float16* __restrict__ packB,
            const float* __restrict__ bu, float* __restrict__ agg, int nNodes) {
  const int wave = (int)((blockIdx.x * blockDim.x + threadIdx.x) >> 5);
  const int lane = (int)(threadIdx.x & 31);
  const int hi = lane >> 4, r = lane & 15;
  const int base = wave * 16;
  if (base >= nNodes) return;

  v16h a = {};  // lanes<16: K0..7; lanes>=16: K8..15
  load8h(h1 + (size_t)(base + r) * H1 + hi * 8, a, 0);

#pragma unroll
  for (int t = 0; t < 2; ++t) {
    const v16h b = load_frag(packB, FRAG_WU2(t), lane);
    v8f c = {};
    c = wmma16(a, b, c);
    const float bias = bu[t * 16 + r];
#pragma unroll
    for (int v = 0; v < 8; ++v) {
      const int m = v + hi * 8;
      agg[(size_t)(base + m) * H2 + t * 16 + r] = c[v] + bias;
    }
  }
}

// ----------------------------------------------------------------------------
// Kernel 4: layer-2 edge messages. z[36] = {h1[dst](16), h1[src](16), ea(4)},
// K padded to 64 => 2 K-steps; out 32 => 2 N-tiles. 8 WMMAs/wave.
// ----------------------------------------------------------------------------
extern "C" __global__ void __launch_bounds__(256)
k_edge2(const float* __restrict__ h1, const float* __restrict__ ea,
        const _Float16* __restrict__ packB,
        const float* __restrict__ bf, const float* __restrict__ bs,
        const int* __restrict__ ei, float* __restrict__ agg, int nEdges) {
  const int wave = (int)((blockIdx.x * blockDim.x + threadIdx.x) >> 5);
  const int lane = (int)(threadIdx.x & 31);
  const int hi = lane >> 4, r = lane & 15;
  const int base = wave * 16;
  if (base >= nEdges) return;
  const int* __restrict__ srcI = ei;
  const int* __restrict__ dstI = ei + nEdges;
  const int e = base + r;

  // A0: K 0..31 (z[0..15]=h1[dst], z[16..31]=h1[src])
  v16h a0 = {};
  load8h(h1 + (size_t)dstI[e] * H1 + hi * 8, a0, 0);
  load8h(h1 + (size_t)srcI[e] * H1 + hi * 8, a0, 8);
  // A1: K 32..63 (z[32..35]=edge_attr; rest zero)
  v16h a1 = {};
  if (hi == 0) {
    const float4 q = *(const float4*)(ea + (size_t)e * EDIM);
    a1[0] = (_Float16)q.x; a1[1] = (_Float16)q.y;
    a1[2] = (_Float16)q.z; a1[3] = (_Float16)q.w;
  }

  v8f cf[2] = {{}, {}}, cs[2] = {{}, {}};
#pragma unroll
  for (int t = 0; t < 2; ++t) {
#pragma unroll
    for (int kb = 0; kb < 2; ++kb) {
      const v16h vbf = load_frag(packB, FRAG_WF2(kb, t), lane);
      const v16h vbs = load_frag(packB, FRAG_WS2(kb, t), lane);
      const v16h aa = kb ? a1 : a0;
      cf[t] = wmma16(aa, vbf, cf[t]);
      cs[t] = wmma16(aa, vbs, cs[t]);
    }
  }

  const float bfv[2] = {bf[r], bf[16 + r]};
  const float bsv[2] = {bs[r], bs[16 + r]};
#pragma unroll
  for (int v = 0; v < 8; ++v) {
    const int m = v + hi * 8;
    const int dd = dstI[base + m];
#pragma unroll
    for (int t = 0; t < 2; ++t) {
      const float val = sigm(cf[t][v] + bfv[t]) * softplusf(cs[t][v] + bsv[t]);
      atomicAdd(&agg[(size_t)dd * H2 + t * 16 + r], val);
    }
  }
}

// ----------------------------------------------------------------------------
// Kernel 5: temporal conv as im2col GEMM. Per wave: 16 time steps of one joint.
// Y[t][c] = relu( sum_{dk,i} cw[c][i][dk]*h2[(max(t-2+dk,0))*J+j][i] + cb[c] )
//           + h2[t*J+j][c]
// K = 96 (3 dk-blocks of 32), N = 32 (2 tiles) -> 6 WMMAs/wave.
// ----------------------------------------------------------------------------
extern "C" __global__ void __launch_bounds__(256)
k_tconv(const float* __restrict__ h2, const _Float16* __restrict__ packB,
        const float* __restrict__ cb, const int* __restrict__ ngPtr,
        float* __restrict__ out, int nNodes) {
  const int wave = (int)((blockIdx.x * blockDim.x + threadIdx.x) >> 5);
  const int lane = (int)(threadIdx.x & 31);
  const int hi = lane >> 4, r = lane & 15;
  const int T = *ngPtr;          // num_graphs (frames)
  const int J = nNodes / T;      // joints
  const int nT16 = T >> 4;
  if (wave >= J * nT16) return;
  const int j = wave / nT16;
  const int tbase = (wave - j * nT16) * 16;

  v8f c0 = {}, c1 = {};
  const int tm = tbase + r;  // matrix row for this lane (both half-waves)
#pragma unroll
  for (int dk = 0; dk < 3; ++dk) {
    int tc = tm - 2 + dk;
    if (tc < 0) tc = 0;  // left replication (edge) padding
    const float* row = h2 + ((size_t)tc * J + j) * H2;
    // lane covers K = {0..7,16..23} (hi=0) or {8..15,24..31} (hi=1)
    v16h a = {};
    load8h(row + hi * 8, a, 0);
    load8h(row + 16 + hi * 8, a, 8);
    const v16h b0 = load_frag(packB, FRAG_CW(dk, 0), lane);
    const v16h b1 = load_frag(packB, FRAG_CW(dk, 1), lane);
    c0 = wmma16(a, b0, c0);
    c1 = wmma16(a, b1, c1);
  }
  const float bias0 = cb[r], bias1 = cb[16 + r];
#pragma unroll
  for (int v = 0; v < 8; ++v) {
    const int m = v + hi * 8;
    const size_t node = (size_t)(tbase + m) * J + j;
    const float* row = h2 + node * H2;
    float* orow = out + node * H2;  // output row order == node order
    orow[r]      = fmaxf(c0[v] + bias0, 0.0f) + row[r];
    orow[16 + r] = fmaxf(c1[v] + bias1, 0.0f) + row[16 + r];
  }
}

// ----------------------------------------------------------------------------
extern "C" void kernel_launch(void* const* d_in, const int* in_sizes, int n_in,
                              void* d_out, int out_size, void* d_ws, size_t ws_size,
                              hipStream_t stream) {
  const float* x   = (const float*)d_in[0];
  const float* ea  = (const float*)d_in[1];
  const float* Wf1 = (const float*)d_in[2];
  const float* bf1 = (const float*)d_in[3];
  const float* Ws1 = (const float*)d_in[4];
  const float* bs1 = (const float*)d_in[5];
  const float* Wu1 = (const float*)d_in[6];
  const float* bu1 = (const float*)d_in[7];
  const float* Wf2 = (const float*)d_in[8];
  const float* bf2 = (const float*)d_in[9];
  const float* Ws2 = (const float*)d_in[10];
  const float* bs2 = (const float*)d_in[11];
  const float* Wu2 = (const float*)d_in[12];
  const float* bu2 = (const float*)d_in[13];
  const float* cw  = (const float*)d_in[14];
  const float* cb  = (const float*)d_in[15];
  const int*   ei  = (const int*)d_in[16];
  const int*   ng  = (const int*)d_in[17];

  const int nNodes = in_sizes[0] / CIN;   // x is [N, 8]
  const int nEdges = in_sizes[1] / EDIM;  // edge_attr is [E, 4]

  _Float16* packB = (_Float16*)d_ws;                         // 19 KB (padded)
  float* agg1 = (float*)((char*)d_ws + PACKB_BYTES);         // [N, 16]
  float* agg2 = agg1 + (size_t)nNodes * H1;                  // [N, 32]

  const int THREADS = 256;                // 8 wave32s per block
  const int WPB = THREADS / 32;
  const int nodeTiles = (nNodes + 15) / 16;
  const int edgeTiles = (nEdges + 15) / 16;
  const int gN = (nodeTiles + WPB - 1) / WPB;
  const int gE = (edgeTiles + WPB - 1) / WPB;

  k_packB    <<<1, 640, 0, stream>>>(Wu1, Wf1, Ws1, Wu2, Wf2, Ws2, cw, packB);
  k_upsample1<<<gN, THREADS, 0, stream>>>(x, packB, bu1, agg1, nNodes);
  k_edge1    <<<gE, THREADS, 0, stream>>>(x, ea, packB, bf1, bs1, ei, agg1, nEdges);
  k_upsample2<<<gN, THREADS, 0, stream>>>(agg1, packB, bu2, agg2, nNodes);
  k_edge2    <<<gE, THREADS, 0, stream>>>(agg1, ea, packB, bf2, bs2, ei, agg2, nEdges);
  k_tconv    <<<gN, THREADS, 0, stream>>>(agg2, packB, cb, ng, (float*)d_out, nNodes);
}
